// InteractionGNNCell_80753975099945
// MI455X (gfx1250) — compile-verified, matrix operand order checked by reading, not run on hardware
//
#include <hip/hip_runtime.h>
#include <hip/hip_bf16.h>
#include <math.h>

typedef __attribute__((ext_vector_type(16))) _Float16 v16h;
typedef __attribute__((ext_vector_type(8)))  _Float16 v8h;
typedef __attribute__((ext_vector_type(8)))  float    v8f;
typedef __attribute__((ext_vector_type(4)))  float    v4f;

#define LN_EPS 1e-5f
#define LATENT 64
#define LDY 132   // f32 accumulator LDS row stride (128+4 pad)

struct LnP8 { float g[8]; float b[8]; };
struct LnP4 { float g[4]; float b[4]; };

// ---------------------------------------------------------------------------
// WMMA fragment helpers (CDNA5 ISA 7.12.2 layouts, wave32)
// A 16x32 f16: lanes 0-15 -> M=lane, K={kb+0..7, kb+16..23};
//              lanes16-31 -> M=lane-16, K={kb+8..15, kb+24..31}.
// Each lane's A fragment = two contiguous 16B runs of its row.
// ---------------------------------------------------------------------------
__device__ __forceinline__ v16h load_a_frag(const _Float16* sX, int ldx, int lane, int kb) {
  const int hf = lane >> 4;
  const int m  = lane & 15;
  const _Float16* p = sX + m * ldx + kb + hf * 8;
  v8h lo = *(const v8h*)(p);
  v8h hi = *(const v8h*)(p + 16);
  return __builtin_shufflevector(lo, hi, 0, 1, 2, 3, 4, 5, 6, 7,
                                 8, 9, 10, 11, 12, 13, 14, 15);
}

// B fragments stored PRE-SWIZZLED in LDS: lane L of fragment f owns 16
// contiguous halves at sW[(f*32+L)*16]  ->  one aligned v16h (2x ds_load_b128).
__device__ __forceinline__ v16h load_b_frag(const _Float16* sW, int frag, int lane) {
  return *(const v16h*)(sW + ((frag * 32 + lane) << 4));
}

// Stage W (f32 global, row-major [CIN][COUT]) into fragment-ordered f16 LDS.
// Done ONCE per block (persistent blocks); each thread emits contiguous v8h.
template <int CIN, int COUT>
__device__ __forceinline__ void stage_w_frag(const float* __restrict__ W,
                                             _Float16* sW, int tid) {
  constexpr int NT    = COUT / 16;
  constexpr int NFRAG = (CIN / 32) * NT;
  for (int u = tid; u < NFRAG * 64; u += 256) {
    const int frag  = u >> 6;
    const int rem   = u & 63;
    const int lane  = rem >> 1;
    const int g     = rem & 1;
    const int chunk = frag / NT;
    const int nt    = frag % NT;
    const int hf    = lane >> 4, n = lane & 15;
    const int c     = nt * 16 + n;
    const int kbase = chunk * 32 + g * 16 + hf * 8;
    v8h t;
#pragma unroll
    for (int j = 0; j < 8; ++j) t[j] = (_Float16)W[(kbase + j) * COUT + c];
    *(v8h*)(sW + ((frag * 32 + lane) << 4) + g * 8) = t;
  }
}

// One wave computes a 16x16 output tile: sY[.][nb..nb+15] = sX @ sW + bias
template <int CIN, int COUT>
__device__ __forceinline__ void wave_gemm_tile(const _Float16* sX, int ldx,
                                               const _Float16* sW, float bv,
                                               float* sY, int lane, int nb) {
  constexpr int NT = COUT / 16;
  const int ntile = nb >> 4;
  v8f c = {};
#pragma unroll
  for (int kc = 0; kc < CIN / 32; ++kc) {
    v16h a = load_a_frag(sX, ldx, lane, kc * 32);
    v16h b = load_b_frag(sW, kc * NT + ntile, lane);
    c = __builtin_amdgcn_wmma_f32_16x16x32_f16(false, a, false, b, (short)0, c, false, false);
  }
  const int hf  = lane >> 4;
  const int col = nb + (lane & 15);
#pragma unroll
  for (int i = 0; i < 8; ++i) sY[(i + 8 * hf) * LDY + col] = c[i] + bv;
}

// LayerNorm + ReLU -> f16 into sX. 16 threads/row own contiguous 8-chunks;
// stats via shfl butterflies within 16-lane halves (wave32).
__device__ __forceinline__ void ln_relu_to_x(const float* sY, const LnP8& p,
                                             _Float16* sX, int ldx, int tid) {
  const int r = tid >> 4, t = tid & 15;
  const float* yrow = sY + r * LDY + t * 8;
  float s = 0.f, sq = 0.f;
#pragma unroll
  for (int j = 0; j < 8; ++j) { float v = yrow[j]; s += v; sq += v * v; }
#pragma unroll
  for (int off = 8; off >= 1; off >>= 1) {
    s  += __shfl_xor(s, off, 16);
    sq += __shfl_xor(sq, off, 16);
  }
  const float mean = s * (1.0f / 128.0f);
  const float inv  = rsqrtf(sq * (1.0f / 128.0f) - mean * mean + LN_EPS);
  v8h o;
#pragma unroll
  for (int j = 0; j < 8; ++j) {
    float v = (yrow[j] - mean) * inv * p.g[j] + p.b[j];
    o[j] = (_Float16)fmaxf(v, 0.f);
  }
  *(v8h*)(sX + r * ldx + t * 8) = o;     // one 16B LDS store
}

// Final layer (C=64): LN + act + residual, one float4 load/store per thread.
template <int ACT>  // 0=ReLU, 1=Tanh
__device__ __forceinline__ void ln_act_residual_store(const float* sY, const LnP4& p,
                                                      const float* __restrict__ resid,
                                                      float* __restrict__ out,
                                                      long long row0, long long nrows,
                                                      int tid) {
  const int r = tid >> 4, t = tid & 15;
  const float* yrow = sY + r * LDY + t * 4;
  float s = 0.f, sq = 0.f;
#pragma unroll
  for (int j = 0; j < 4; ++j) { float v = yrow[j]; s += v; sq += v * v; }
#pragma unroll
  for (int off = 8; off >= 1; off >>= 1) {
    s  += __shfl_xor(s, off, 16);
    sq += __shfl_xor(sq, off, 16);
  }
  const float mean = s * (1.0f / 64.0f);
  const float inv  = rsqrtf(sq * (1.0f / 64.0f) - mean * mean + LN_EPS);
  const long long row = row0 + r;
  if (row < nrows) {
    const v4f rv = *(const v4f*)(resid + row * 64 + t * 4);
    v4f o;
#pragma unroll
    for (int j = 0; j < 4; ++j) {
      float v = (yrow[j] - mean) * inv * p.g[j] + p.b[j];
      v = (ACT == 0) ? fmaxf(v, 0.f) : tanhf(v);
      o[j] = v + rv[j];
    }
    *(v4f*)(out + row * 64 + t * 4) = o;
  }
}

// ---------------------------------------------------------------------------
// K1: zero messages
// ---------------------------------------------------------------------------
__global__ void zero_f32_kernel(float* __restrict__ p, long long n) {
  long long i = (long long)blockIdx.x * 256 + threadIdx.x;
  if (i < n) p[i] = 0.f;
}

// ---------------------------------------------------------------------------
// K2: scatter-add edge features into messages[dst]. Coalesced edge reads,
// L2-resident atomics (messages = 25.6 MB << 192 MB L2).
// ---------------------------------------------------------------------------
__global__ void scatter_add_kernel(const float* __restrict__ edges,
                                   const int* __restrict__ dst,
                                   float* __restrict__ msgs, long long n_edges) {
  long long i = (long long)blockIdx.x * 256 + threadIdx.x;
  if (i >= n_edges * LATENT) return;
  const long long e = i >> 6;
  const int f = (int)(i & 63);
  atomicAdd(&msgs[(long long)dst[e] * LATENT + f], edges[i]);
}

// ---------------------------------------------------------------------------
// K3: node MLP  [nodes|messages](128) -> 128 -> 128 -> 64 (+residual)
// Persistent blocks: weights staged to LDS once, grid-stride over 16-row
// tiles. 256 threads = 8 waves; wave w owns output cols 16w..16w+15.
// ---------------------------------------------------------------------------
#define NLDX 136
__global__ __launch_bounds__(256) void node_mlp_kernel(
    const float* __restrict__ nodes, const float* __restrict__ msgs,
    const float* __restrict__ W0, const float* __restrict__ b0,
    const float* __restrict__ g0, const float* __restrict__ be0,
    const float* __restrict__ W1, const float* __restrict__ b1,
    const float* __restrict__ g1, const float* __restrict__ be1,
    const float* __restrict__ W2, const float* __restrict__ b2,
    const float* __restrict__ g2, const float* __restrict__ be2,
    float* __restrict__ out, long long n_nodes) {
  __shared__ __align__(32) _Float16 sX[16 * NLDX];
  __shared__ __align__(16) float    sY[16 * LDY];
  __shared__ __align__(32) _Float16 sW0[128 * 128];
  __shared__ __align__(32) _Float16 sW1[128 * 128];
  __shared__ __align__(32) _Float16 sW2[128 * 64];

  const int tid  = threadIdx.x;
  const int wave = tid >> 5, lane = tid & 31;
  const int t    = tid & 15;
  const int col  = wave * 16 + (lane & 15);

  // per-thread constant parameters (indices fixed across tiles)
  LnP8 p0, p1; LnP4 p2;
#pragma unroll
  for (int j = 0; j < 8; ++j) {
    p0.g[j] = g0[t * 8 + j]; p0.b[j] = be0[t * 8 + j];
    p1.g[j] = g1[t * 8 + j]; p1.b[j] = be1[t * 8 + j];
  }
#pragma unroll
  for (int j = 0; j < 4; ++j) { p2.g[j] = g2[t * 4 + j]; p2.b[j] = be2[t * 4 + j]; }
  const float bv0 = b0[col];
  const float bv1 = b1[col];
  const float bv2 = (wave < 4) ? b2[col] : 0.f;

  stage_w_frag<128, 128>(W0, sW0, tid);
  stage_w_frag<128, 128>(W1, sW1, tid);
  stage_w_frag<128, 64>(W2, sW2, tid);
  __syncthreads();

  const long long ntiles = (n_nodes + 15) >> 4;
  for (long long tile = blockIdx.x; tile < ntiles; tile += gridDim.x) {
    const long long row0 = tile * 16;

    // input tile: concat(nodes, messages) -> f16 LDS (float2 per thread)
    for (int i = tid; i < 16 * 64; i += 256) {
      const int r = i >> 6, f = (i & 63) * 2;
      const long long row = row0 + r;
      float2 v = make_float2(0.f, 0.f);
      if (row < n_nodes)
        v = (f < LATENT) ? *(const float2*)(nodes + row * LATENT + f)
                         : *(const float2*)(msgs + row * LATENT + (f - LATENT));
      sX[r * NLDX + f]     = (_Float16)v.x;
      sX[r * NLDX + f + 1] = (_Float16)v.y;
    }
    __syncthreads();

    wave_gemm_tile<128, 128>(sX, NLDX, sW0, bv0, sY, lane, wave * 16);
    __syncthreads();
    ln_relu_to_x(sY, p0, sX, NLDX, tid);
    __syncthreads();

    wave_gemm_tile<128, 128>(sX, NLDX, sW1, bv1, sY, lane, wave * 16);
    __syncthreads();
    ln_relu_to_x(sY, p1, sX, NLDX, tid);
    __syncthreads();

    if (wave < 4) wave_gemm_tile<128, 64>(sX, NLDX, sW2, bv2, sY, lane, wave * 16);
    __syncthreads();
    ln_act_residual_store<0>(sY, p2, nodes, out, row0, n_nodes, tid);
  }
}

// ---------------------------------------------------------------------------
// K4: edge MLP  [n_new[src]|n_new[dst]|edges](192) -> 128 -> 128 -> 64 (+res)
// Persistent blocks; gathers of nodes_new hit L2 (25.6 MB table).
// ---------------------------------------------------------------------------
#define ELDX 200
__global__ __launch_bounds__(256) void edge_mlp_kernel(
    const float* __restrict__ nodes_new, const float* __restrict__ edges,
    const int* __restrict__ src, const int* __restrict__ dst,
    const float* __restrict__ W0, const float* __restrict__ b0,
    const float* __restrict__ g0, const float* __restrict__ be0,
    const float* __restrict__ W1, const float* __restrict__ b1,
    const float* __restrict__ g1, const float* __restrict__ be1,
    const float* __restrict__ W2, const float* __restrict__ b2,
    const float* __restrict__ g2, const float* __restrict__ be2,
    float* __restrict__ out, long long n_edges) {
  __shared__ __align__(32) _Float16 sX[16 * ELDX];
  __shared__ __align__(16) float    sY[16 * LDY];
  __shared__ __align__(32) _Float16 sW0[192 * 128];
  __shared__ __align__(32) _Float16 sW1[128 * 128];
  __shared__ __align__(32) _Float16 sW2[128 * 64];

  const int tid  = threadIdx.x;
  const int wave = tid >> 5, lane = tid & 31;
  const int t    = tid & 15;
  const int col  = wave * 16 + (lane & 15);

  LnP8 p0, p1; LnP4 p2;
#pragma unroll
  for (int j = 0; j < 8; ++j) {
    p0.g[j] = g0[t * 8 + j]; p0.b[j] = be0[t * 8 + j];
    p1.g[j] = g1[t * 8 + j]; p1.b[j] = be1[t * 8 + j];
  }
#pragma unroll
  for (int j = 0; j < 4; ++j) { p2.g[j] = g2[t * 4 + j]; p2.b[j] = be2[t * 4 + j]; }
  const float bv0 = b0[col];
  const float bv1 = b1[col];
  const float bv2 = (wave < 4) ? b2[col] : 0.f;

  stage_w_frag<192, 128>(W0, sW0, tid);
  stage_w_frag<128, 128>(W1, sW1, tid);
  stage_w_frag<128, 64>(W2, sW2, tid);
  __syncthreads();

  const long long ntiles = (n_edges + 15) >> 4;
  for (long long tile = blockIdx.x; tile < ntiles; tile += gridDim.x) {
    const long long row0 = tile * 16;

    // input tile: [n_new[src] | n_new[dst] | edges]; float2 pairs never
    // straddle a segment boundary (boundaries at 64/128 are even).
    for (int i = tid; i < 16 * 96; i += 256) {
      const int r = i / 96, f = (i % 96) * 2;
      const long long e = row0 + r;
      float2 v = make_float2(0.f, 0.f);
      if (e < n_edges) {
        if (f < 64)       v = *(const float2*)(nodes_new + (long long)src[e] * LATENT + f);
        else if (f < 128) v = *(const float2*)(nodes_new + (long long)dst[e] * LATENT + (f - 64));
        else              v = *(const float2*)(edges + e * LATENT + (f - 128));
      }
      sX[r * ELDX + f]     = (_Float16)v.x;
      sX[r * ELDX + f + 1] = (_Float16)v.y;
    }
    __syncthreads();

    wave_gemm_tile<192, 128>(sX, ELDX, sW0, bv0, sY, lane, wave * 16);
    __syncthreads();
    ln_relu_to_x(sY, p0, sX, ELDX, tid);
    __syncthreads();

    wave_gemm_tile<128, 128>(sX, ELDX, sW1, bv1, sY, lane, wave * 16);
    __syncthreads();
    ln_relu_to_x(sY, p1, sX, ELDX, tid);
    __syncthreads();

    if (wave < 4) wave_gemm_tile<128, 64>(sX, ELDX, sW2, bv2, sY, lane, wave * 16);
    __syncthreads();
    ln_act_residual_store<1>(sY, p2, edges, out, row0, n_edges, tid);
  }
}

// ---------------------------------------------------------------------------
extern "C" void kernel_launch(void* const* d_in, const int* in_sizes, int n_in,
                              void* d_out, int out_size, void* d_ws, size_t ws_size,
                              hipStream_t stream) {
  const float* nodes = (const float*)d_in[0];
  const float* edges = (const float*)d_in[1];
  const int*   graph = (const int*)d_in[2];

  const float* eW0 = (const float*)d_in[3];
  const float* eb0 = (const float*)d_in[4];
  const float* eg0 = (const float*)d_in[5];
  const float* ee0 = (const float*)d_in[6];
  const float* eW1 = (const float*)d_in[7];
  const float* eb1 = (const float*)d_in[8];
  const float* eg1 = (const float*)d_in[9];
  const float* ee1 = (const float*)d_in[10];
  const float* eW2 = (const float*)d_in[11];
  const float* eb2 = (const float*)d_in[12];
  const float* eg2 = (const float*)d_in[13];
  const float* ee2 = (const float*)d_in[14];
  const float* nW0 = (const float*)d_in[15];
  const float* nb0 = (const float*)d_in[16];
  const float* ng0 = (const float*)d_in[17];
  const float* ne0 = (const float*)d_in[18];
  const float* nW1 = (const float*)d_in[19];
  const float* nb1 = (const float*)d_in[20];
  const float* ng1 = (const float*)d_in[21];
  const float* ne1 = (const float*)d_in[22];
  const float* nW2 = (const float*)d_in[23];
  const float* nb2 = (const float*)d_in[24];
  const float* ng2 = (const float*)d_in[25];
  const float* ne2 = (const float*)d_in[26];

  const long long N = in_sizes[0] / LATENT;
  const long long E = in_sizes[1] / LATENT;

  float* msgs      = (float*)d_ws;               // [N,64] f32
  float* out_nodes = (float*)d_out;              // [N,64]
  float* out_edges = out_nodes + N * LATENT;     // [E,64]
  const int* src = graph;
  const int* dst = graph + E;

  {
    const long long n = N * LATENT;
    zero_f32_kernel<<<(unsigned)((n + 255) / 256), 256, 0, stream>>>(msgs, n);
  }
  {
    const long long n = E * LATENT;
    scatter_add_kernel<<<(unsigned)((n + 255) / 256), 256, 0, stream>>>(edges, dst, msgs, E);
  }
  {
    const long long ntiles = (N + 15) >> 4;
    const unsigned nblk = (unsigned)(ntiles < 1024 ? ntiles : 1024);
    node_mlp_kernel<<<nblk, 256, 0, stream>>>(
        nodes, msgs, nW0, nb0, ng0, ne0, nW1, nb1, ng1, ne1, nW2, nb2, ng2, ne2,
        out_nodes, N);
  }
  {
    const long long ntiles = (E + 15) >> 4;
    const unsigned nblk = (unsigned)(ntiles < 2048 ? ntiles : 2048);
    edge_mlp_kernel<<<nblk, 256, 0, stream>>>(
        out_nodes, edges, src, dst, eW0, eb0, eg0, ee0, eW1, eb1, eg1, ee1,
        eW2, eb2, eg2, ee2, out_edges, E);
  }
}
